// ILmodel_3762391351958
// MI455X (gfx1250) — matrix operand order repaired and checked by reading, not compiled, not run on hardware
//
#include <hip/hip_runtime.h>
#include <math.h>

// ---------------------------------------------------------------------------
// CDNA5 (gfx1250, wave32) fused MLP-VAE forward.
// Each wave owns a 16-row tile. All GEMMs run on v_wmma_f32_16x16x32_f16.
// Weights are staged once per block into LDS, converted f32->f16 and
// pre-swizzled into the WMMA B-operand layout. Activations bounce through a
// per-wave LDS tile (16 x 96 f16) to convert D-layout (f32) -> A-layout (f16).
// Transcendentals use branch-free hw ops (v_exp_f32 / v_rcp_f32 / v_sqrt_f32).
// ---------------------------------------------------------------------------

typedef _Float16 half_t;
typedef __attribute__((ext_vector_type(16))) _Float16 v16h;
typedef __attribute__((ext_vector_type(2)))  _Float16 h2;
typedef __attribute__((ext_vector_type(8)))  float    v8f;

#define WAVES 8
#define TPB   (WAVES * 32)
#define LDK   96            // per-wave activation staging stride (halfs)

// ---- LDS layout (dynamic shared; ~83 KB, fine for 320KB-per-WGP CDNA5) ----
#define W_HALFS   29184
#define B_FLOATS  416
#define OFF_B     (W_HALFS * 2)                 // 58368
#define OFF_STG   (OFF_B + B_FLOATS * 4)        // 60032 (32B aligned)
#define SMEM_BYTES (OFF_STG + WAVES * 16 * LDK * 2)   // 84608

// weight sub-offsets in halfs ([ntile][kchunk][512] each, 512 = 32x16 halfs)
enum {
  W1_OFF = 0,      // 96x64  : KC=3 NT=4 -> 6144
  W2_OFF = 6144,   // 64x64  : KC=2 NT=4 -> 4096
  W3_OFF = 10240,  // 64x80  : KC=2 NT=5 -> 5120
  C1_OFF = 15360,  // 96x64  : KC=3 NT=4 -> 6144
  C2_OFF = 21504,  // 64x64  : KC=2 NT=4 -> 4096
  C3_OFF = 25600,  // 64x32  : KC=2 NT=2 -> 2048
  C4_OFF = 27648,  // 32x16  : KC=1 NT=1 -> 512
  MW_OFF = 28160,  // 32x16  (16 rows zero-padded)
  LW_OFF = 28672
};
enum {
  B1_OFF = 0, B2_OFF = 64, B3_OFF = 128, CB1_OFF = 208, CB2_OFF = 272,
  CB3_OFF = 336, CB4_OFF = 368, MB_OFF = 384, LB_OFF = 400
};

extern __shared__ char smem[];

// ---- branch-free hardware transcendentals -------------------------------
#define LOG2E_F 1.44269504088896340736f

__device__ __forceinline__ float fast_exp(float x) {
  return __builtin_amdgcn_exp2f(x * LOG2E_F);            // v_exp_f32
}
// tanh(x) = 1 - 2/(exp(2x)+1); saturates to +-1 via IEEE inf/0, no branches.
__device__ __forceinline__ float fast_tanh(float x) {
  float e = __builtin_amdgcn_exp2f(x * (2.0f * LOG2E_F));
  return 1.0f - 2.0f * __builtin_amdgcn_rcpf(e + 1.0f);  // v_rcp_f32
}

// Convert + swizzle a (K x Nn) row-major f32 weight matrix into LDS f16 in
// WMMA B-operand layout, zero-padded to (Kp x Np).
// B staged in contiguous k order (k = 16h + s), same as A: lane = (k%32)/16*16 + n, halfs[ k%16 ] of that lane.
template <int K, int Nn, int Kp, int Np>
__device__ __forceinline__ void stage_w(const float* __restrict__ w,
                                        half_t* dst, int tid) {
  constexpr int KC = Kp >> 5;
  constexpr int total = Kp * Np;
  for (int idx = tid; idx < total; idx += TPB) {
    int k = idx / Np, n = idx % Np;
    float v = (k < K && n < Nn) ? w[k * Nn + n] : 0.0f;
    int nt = n >> 4, nn = n & 15;
    int c  = k >> 5, kin = k & 31;
    int lane = ((kin >> 4) << 4) | nn;
    dst[(nt * KC + c) * 512 + lane * 16 + (kin & 15)] = (half_t)v;
  }
}

template <int Nn, int Np>
__device__ __forceinline__ void stage_b(const float* __restrict__ b,
                                        float* dst, int tid) {
  for (int i = tid; i < Np; i += TPB) dst[i] = (i < Nn) ? b[i] : 0.0f;
}

// One GEMM layer on WMMA: D(16 x NT*16) = A(16 x KC*32) * B + bias.
// A gathered from row-major f16 staging in contiguous k order (same as B staging):
//   lane l: row m = l&15, half = l>>4; vgpr j holds K = half*16 + j*2 (+e)
template <int KC, int NT>
__device__ __forceinline__ void mlp_layer(const half_t* __restrict__ Asrc,
                                          const half_t* __restrict__ Bw,
                                          const float* __restrict__ bias,
                                          v8f* acc, int lane) {
  const int m  = lane & 15;
  const int hs = lane >> 4;
#pragma unroll
  for (int nt = 0; nt < NT; ++nt) {
    float bv = bias[nt * 16 + m];
#pragma unroll
    for (int j = 0; j < 8; ++j) acc[nt][j] = bv;
  }
#pragma unroll
  for (int c = 0; c < KC; ++c) {
    v16h a;
#pragma unroll
    for (int j = 0; j < 8; ++j) {
      int k = c * 32 + (hs << 4) + (j << 1);
      h2 p = *(const h2*)(Asrc + m * LDK + k);
      a[2 * j]     = p[0];
      a[2 * j + 1] = p[1];
    }
#pragma unroll
    for (int nt = 0; nt < NT; ++nt) {
      v16h b = *(const v16h*)(Bw + (nt * KC + c) * 512 + lane * 16);
      acc[nt] = __builtin_amdgcn_wmma_f32_16x16x32_f16(
          false, a, false, b, (short)0, acc[nt], false, false);
    }
  }
}

// D layout (f32 16x16): lanes 0-15 vgpr j -> M=j, N=lane; lanes 16-31 -> M=j+8.
// ACT: 0 = none, 1 = tanh, 2 = relu. Store back to f16 staging, row-major.
template <int ACT>
__device__ __forceinline__ void store_act(half_t* dst, const v8f& a, int nt, int lane) {
  const int n  = (lane & 15) + nt * 16;
  const int mb = (lane >> 4) * 8;
#pragma unroll
  for (int j = 0; j < 8; ++j) {
    float v = a[j];
    if (ACT == 1) v = fast_tanh(v);
    else if (ACT == 2) v = fmaxf(v, 0.0f);
    dst[(mb + j) * LDK + n] = (half_t)v;
  }
}

__global__ void __launch_bounds__(TPB)
mlp_vae_kernel(const float* __restrict__ obs, const float* __restrict__ yin,
               const float* __restrict__ eps,
               const float* __restrict__ ew1, const float* __restrict__ eb1,
               const float* __restrict__ ew2, const float* __restrict__ eb2,
               const float* __restrict__ ew3, const float* __restrict__ eb3,
               const float* __restrict__ cw1, const float* __restrict__ cb1,
               const float* __restrict__ cw2, const float* __restrict__ cb2,
               const float* __restrict__ cw3, const float* __restrict__ cb3,
               const float* __restrict__ cw4, const float* __restrict__ cb4,
               const float* __restrict__ mw,  const float* __restrict__ mb,
               const float* __restrict__ lw,  const float* __restrict__ lb,
               float* __restrict__ out, int n_rows) {
  half_t* sW = (half_t*)smem;
  float*  sB = (float*)(smem + OFF_B);
  const int tid = threadIdx.x;

  // ---- stage all weights/biases (once per block) ----
  stage_w<76, 64, 96, 64>(ew1, sW + W1_OFF, tid);
  stage_w<64, 64, 64, 64>(ew2, sW + W2_OFF, tid);
  stage_w<64, 76, 64, 80>(ew3, sW + W3_OFF, tid);
  stage_w<76, 64, 96, 64>(cw1, sW + C1_OFF, tid);
  stage_w<64, 64, 64, 64>(cw2, sW + C2_OFF, tid);
  stage_w<64, 32, 64, 32>(cw3, sW + C3_OFF, tid);
  stage_w<32, 16, 32, 16>(cw4, sW + C4_OFF, tid);
  stage_w<16, 16, 32, 16>(mw,  sW + MW_OFF, tid);
  stage_w<16, 16, 32, 16>(lw,  sW + LW_OFF, tid);
  stage_b<64, 64>(eb1, sB + B1_OFF,  tid);
  stage_b<64, 64>(eb2, sB + B2_OFF,  tid);
  stage_b<76, 80>(eb3, sB + B3_OFF,  tid);
  stage_b<64, 64>(cb1, sB + CB1_OFF, tid);
  stage_b<64, 64>(cb2, sB + CB2_OFF, tid);
  stage_b<32, 32>(cb3, sB + CB3_OFF, tid);
  stage_b<16, 16>(cb4, sB + CB4_OFF, tid);
  stage_b<16, 16>(mb,  sB + MB_OFF,  tid);
  stage_b<16, 16>(lb,  sB + LB_OFF,  tid);
  __syncthreads();

  const int wave = tid >> 5;
  const int lane = tid & 31;
  half_t* stg = (half_t*)(smem + OFF_STG) + wave * 16 * LDK;

  const int ntiles = n_rows >> 4;
  const int stride = gridDim.x * WAVES;
  const size_t plane = (size_t)n_rows * 16;

  for (int tile = blockIdx.x * WAVES + wave; tile < ntiles; tile += stride) {
    const int row0 = tile << 4;

    // Prefetch next tile's obs rows (global_prefetch_b8); 32 lanes x 128B = 4KB
    {
      int nxt = tile + stride;
      if (nxt < ntiles)
        __builtin_prefetch(obs + (size_t)(nxt << 4) * 64 + lane * 32, 0, 1);
    }

    // ---- build x = [obs[:,4:64] | y | zero-pad to 96] as f16 tile ----
    for (int i = lane; i < 16 * 20; i += 32) {         // zero cols 76..95
      int r = i / 20, c = 76 + i % 20;
      stg[r * LDK + c] = (half_t)0.0f;
    }
    const float4* obs4 = (const float4*)(obs + (size_t)row0 * 64);
#pragma unroll
    for (int i = 0; i < 8; ++i) {                      // 16x64 obs floats
      int idx = i * 32 + lane;
      float4 v = obs4[idx];
      int r = idx >> 4, c0 = (idx & 15) << 2;
#pragma unroll
      for (int q = 0; q < 4; ++q) {
        int c = c0 + q;
        if (c >= 4) stg[r * LDK + (c - 4)] = (half_t)((&v.x)[q]);
      }
    }
    const float4* y4 = (const float4*)(yin + (size_t)row0 * 16);
#pragma unroll
    for (int i = 0; i < 2; ++i) {                      // 16x16 y floats
      int idx = i * 32 + lane;
      float4 v = y4[idx];
      int r = idx >> 2, c0 = (idx & 3) << 2;
#pragma unroll
      for (int q = 0; q < 4; ++q)
        stg[r * LDK + 60 + c0 + q] = (half_t)((&v.x)[q]);
    }

    v8f acc[5];
    // encoder: 76(->96) x 64, tanh
    mlp_layer<3, 4>(stg, sW + W1_OFF, sB + B1_OFF, acc, lane);
#pragma unroll
    for (int nt = 0; nt < 4; ++nt) store_act<1>(stg, acc[nt], nt, lane);
    // 64 x 64, tanh
    mlp_layer<2, 4>(stg, sW + W2_OFF, sB + B2_OFF, acc, lane);
#pragma unroll
    for (int nt = 0; nt < 4; ++nt) store_act<1>(stg, acc[nt], nt, lane);
    // 64 x 76(->80), linear (pad cols 76..79 come out exactly 0)
    mlp_layer<2, 5>(stg, sW + W3_OFF, sB + B3_OFF, acc, lane);
#pragma unroll
    for (int nt = 0; nt < 5; ++nt) store_act<0>(stg, acc[nt], nt, lane);
    // controller: 76(->96) x 64, relu (cols 80..95 still zero from x build)
    mlp_layer<3, 4>(stg, sW + C1_OFF, sB + CB1_OFF, acc, lane);
#pragma unroll
    for (int nt = 0; nt < 4; ++nt) store_act<2>(stg, acc[nt], nt, lane);
    // 64 x 64, relu
    mlp_layer<2, 4>(stg, sW + C2_OFF, sB + CB2_OFF, acc, lane);
#pragma unroll
    for (int nt = 0; nt < 4; ++nt) store_act<2>(stg, acc[nt], nt, lane);
    // 64 x 32, relu
    mlp_layer<2, 2>(stg, sW + C3_OFF, sB + CB3_OFF, acc, lane);
#pragma unroll
    for (int nt = 0; nt < 2; ++nt) store_act<2>(stg, acc[nt], nt, lane);
    // 32 x 16, linear -> z
    mlp_layer<1, 1>(stg, sW + C4_OFF, sB + CB4_OFF, acc, lane);
    store_act<0>(stg, acc[0], 0, lane);
    // zero cols 16..31 so z is K-padded for the 16->16 heads
    for (int i = lane; i < 256; i += 32) {
      int r = i >> 4, c = 16 + (i & 15);
      stg[r * LDK + c] = (half_t)0.0f;
    }

    v8f mu_a, sd_a;
    mlp_layer<1, 1>(stg, sW + MW_OFF, sB + MB_OFF, &mu_a, lane);
    mlp_layer<1, 1>(stg, sW + LW_OFF, sB + LB_OFF, &sd_a, lane);

    // ---- elementwise heads + reparameterized sample, stores in D layout ----
    const int n   = lane & 15;
    const int mbs = (lane >> 4) * 8;
#pragma unroll
    for (int j = 0; j < 8; ++j) {
      size_t r = (size_t)row0 + mbs + j;
      size_t o = r * 16 + n;
      float e  = eps[o];
      float mu = fast_tanh(fminf(fmaxf(mu_a[j], -7.24f), 7.24f));
      float sd = fast_exp(fminf(fmaxf(sd_a[j], -5.0f), 2.0f));
      out[o]             = mu + __builtin_amdgcn_sqrtf(sd) * e;  // z_sample
      out[plane + o]     = mu;                                   // z_mu
      out[2 * plane + o] = sd;                                   // z_std
    }
  }
}

extern "C" void kernel_launch(void* const* d_in, const int* in_sizes, int n_in,
                              void* d_out, int out_size, void* d_ws, size_t ws_size,
                              hipStream_t stream) {
  (void)n_in; (void)d_ws; (void)ws_size; (void)out_size;
  const float* obs = (const float*)d_in[0];
  const float* y   = (const float*)d_in[1];
  const float* eps = (const float*)d_in[2];
  const float* ew1 = (const float*)d_in[3];  const float* eb1 = (const float*)d_in[4];
  const float* ew2 = (const float*)d_in[5];  const float* eb2 = (const float*)d_in[6];
  const float* ew3 = (const float*)d_in[7];  const float* eb3 = (const float*)d_in[8];
  const float* cw1 = (const float*)d_in[9];  const float* cb1 = (const float*)d_in[10];
  const float* cw2 = (const float*)d_in[11]; const float* cb2 = (const float*)d_in[12];
  const float* cw3 = (const float*)d_in[13]; const float* cb3 = (const float*)d_in[14];
  const float* cw4 = (const float*)d_in[15]; const float* cb4 = (const float*)d_in[16];
  const float* mw  = (const float*)d_in[17]; const float* mb  = (const float*)d_in[18];
  const float* lw  = (const float*)d_in[19]; const float* lb  = (const float*)d_in[20];

  const int n_rows = in_sizes[0] / 64;       // obs is (N, 64)
  const int ntiles = n_rows / 16;
  int blocks = (ntiles + WAVES - 1) / WAVES;
  if (blocks > 2048) blocks = 2048;

  mlp_vae_kernel<<<blocks, TPB, SMEM_BYTES, stream>>>(
      obs, y, eps, ew1, eb1, ew2, eb2, ew3, eb3,
      cw1, cb1, cw2, cb2, cw3, cb3, cw4, cb4,
      mw, mb, lw, lb, (float*)d_out, n_rows);
}
